// TransformerBlock_77232101917092
// MI455X (gfx1250) — matrix operand order repaired and checked by reading, not compiled, not run on hardware
//
#include <hip/hip_runtime.h>

// MI455X / gfx1250, wave32. Transformer block with the ctx[:, -1] dead-code
// exploited: attention collapses to rank-1 -> only the MLP GEMMs are heavy,
// done with v_wmma_f32_16x16x32_bf16 (fp32 accumulate) fed by
// global_load_async_to_lds_b128 double-buffered pipelines.

typedef __attribute__((ext_vector_type(16))) __bf16 v16bf;
typedef __attribute__((ext_vector_type(8)))  float v8f;

constexpr int T = 2048, D = 2048, H = 16, HD = 128, D4 = 8192;

union FragBF { v16bf v; uint4 q[2]; };

__device__ __forceinline__ float warp_sum(float v) {
  for (int o = 16; o > 0; o >>= 1) v += __shfl_xor(v, o, 32);
  return v;
}

__device__ __forceinline__ float gelu_tanh(float x) {
  const float c = 0.7978845608028654f; // sqrt(2/pi)
  return 0.5f * x * (1.0f + tanhf(c * (x + 0.044715f * x * x * x)));
}

// ---- CDNA5 async copy: global -> LDS, 16B, tracked by ASYNCcnt ----
__device__ __forceinline__ void async_ld_b128(const __bf16* gsrc, __bf16* ldst) {
  uint32_t lds = (uint32_t)(uintptr_t)ldst;   // low 32 bits = LDS byte address
  uint64_t ga  = (uint64_t)(uintptr_t)gsrc;
  asm volatile("global_load_async_to_lds_b128 %0, %1, off"
               :: "v"(lds), "v"(ga) : "memory");
}
__device__ __forceinline__ void wait_async0() {
  asm volatile("s_wait_asynccnt 0x0" ::: "memory");
}

// ---------------- LayerNorm (optionally +broadcast vector, residual out, bf16 out)
__global__ __launch_bounds__(256) void ln_kernel(
    const float* __restrict__ xin, const float* __restrict__ addv,
    const float* __restrict__ sc, const float* __restrict__ sh,
    float* __restrict__ resOut, float* __restrict__ f32Out,
    __bf16* __restrict__ bfOut) {
  const int t = blockIdx.x, tid = threadIdx.x;
  const size_t base = (size_t)t * D;
  float v[8];
  float s = 0.f, ss = 0.f;
#pragma unroll
  for (int i = 0; i < 8; ++i) {
    int j = tid + i * 256;
    float val = xin[base + j];
    if (addv) val += addv[j];
    v[i] = val; s += val; ss += val * val;
  }
  __shared__ float red[16];
  s = warp_sum(s); ss = warp_sum(ss);
  int w = tid >> 5, l = tid & 31;
  if (l == 0) { red[w] = s; red[8 + w] = ss; }
  __syncthreads();
  float S0 = 0.f, S1 = 0.f;
#pragma unroll
  for (int i = 0; i < 8; ++i) { S0 += red[i]; S1 += red[8 + i]; }
  const float mean = S0 * (1.0f / D);
  const float var  = S1 * (1.0f / D) - mean * mean;
  const float inv  = rsqrtf(var + 1e-5f);
#pragma unroll
  for (int i = 0; i < 8; ++i) {
    int j = tid + i * 256;
    if (resOut) resOut[base + j] = v[i];
    float y = sc[j] * (v[i] - mean) * inv + sh[j];
    if (f32Out) f32Out[base + j] = y;
    if (bfOut)  bfOut[base + j]  = (__bf16)y;
  }
}

// ---------------- out[j] = sum_i v[i]*W[i*N+j] (+bias) ; coalesced over j
__global__ __launch_bounds__(256) void matvec_kernel(
    const float* __restrict__ v, const float* __restrict__ W,
    const float* __restrict__ bias, float* __restrict__ out, int K, int N) {
  int j = blockIdx.x * 256 + threadIdx.x;
  float acc = bias ? bias[j] : 0.0f;
  for (int i = 0; i < K; ++i) acc += v[i] * W[(size_t)i * N + j];
  out[j] = acc;
}

// ---------------- g[h][i] = Wk[i, h*128 : h*128+128] . q[h*128 : +128]
__global__ __launch_bounds__(256) void gvec_kernel(
    const float* __restrict__ Wk, const float* __restrict__ q,
    float* __restrict__ g) {
  int pair = blockIdx.x * 8 + (threadIdx.x >> 5);   // H*D pairs, one per wave
  int lane = threadIdx.x & 31;
  int h = pair >> 11;          // / D
  int i = pair & (D - 1);
  const float4* wrow = reinterpret_cast<const float4*>(Wk + (size_t)i * D + h * HD);
  const float4* qh   = reinterpret_cast<const float4*>(q + h * HD);
  float4 a = wrow[lane], b = qh[lane];
  float acc = a.x * b.x + a.y * b.y + a.z * b.z + a.w * b.w;
  acc = warp_sum(acc);
  if (lane == 0) g[(size_t)h * D + i] = acc;
}

// ---------------- S[t][h] = (h1[t] . g[h]) / sqrt(HD)   (16 waves = 16 heads)
__global__ __launch_bounds__(512) void scores_kernel(
    const float* __restrict__ h1, const float* __restrict__ g,
    float* __restrict__ S) {
  __shared__ float hs[D];
  const int t = blockIdx.x, tid = threadIdx.x;
  const float* row = h1 + (size_t)t * D;
  for (int i = tid; i < D; i += 512) hs[i] = row[i];
  __syncthreads();
  int w = tid >> 5, lane = tid & 31;
  const float* gh = g + (size_t)w * D;
  float acc = 0.f;
  for (int i = lane; i < D; i += 32) acc += hs[i] * gh[i];
  acc = warp_sum(acc);
  if (lane == 0) S[(size_t)t * H + w] = acc * 0.08838834764831845f; // 1/sqrt(128)
}

// ---------------- softmax over t for each head, in place
__global__ __launch_bounds__(256) void softmax_kernel(float* __restrict__ S) {
  const int h = blockIdx.x, tid = threadIdx.x;
  __shared__ float red[8];
  float vals[8];
  float m = -3.402823466e38f;
#pragma unroll
  for (int i = 0; i < 8; ++i) {
    vals[i] = S[(size_t)(tid + i * 256) * H + h];
    m = fmaxf(m, vals[i]);
  }
  for (int o = 16; o > 0; o >>= 1) m = fmaxf(m, __shfl_xor(m, o, 32));
  int w = tid >> 5, l = tid & 31;
  if (l == 0) red[w] = m;
  __syncthreads();
  float mm = red[0];
#pragma unroll
  for (int i = 1; i < 8; ++i) mm = fmaxf(mm, red[i]);
  float s = 0.f;
#pragma unroll
  for (int i = 0; i < 8; ++i) { vals[i] = expf(vals[i] - mm); s += vals[i]; }
  s = warp_sum(s);
  __syncthreads();
  if (l == 0) red[w] = s;
  __syncthreads();
  float tot = 0.f;
#pragma unroll
  for (int i = 0; i < 8; ++i) tot += red[i];
  float inv = 1.0f / tot;
#pragma unroll
  for (int i = 0; i < 8; ++i) S[(size_t)(tid + i * 256) * H + h] = vals[i] * inv;
}

// ---------------- M[h][j] = sum_t A[t][h] * h1[t][j]  (A^T @ h1, single h1 pass)
__global__ __launch_bounds__(256) void wsum_kernel(
    const float* __restrict__ A, const float* __restrict__ h1,
    float* __restrict__ Mh) {
  const int j = blockIdx.x * 256 + threadIdx.x;
  __shared__ float As[128 * H];
  float acc[H];
#pragma unroll
  for (int h = 0; h < H; ++h) acc[h] = 0.f;
  for (int t0 = 0; t0 < T; t0 += 128) {
    for (int i = threadIdx.x; i < 128 * H; i += 256) As[i] = A[(size_t)t0 * H + i];
    __syncthreads();
    for (int tt = 0; tt < 128; ++tt) {
      float hv = h1[(size_t)(t0 + tt) * D + j];
#pragma unroll
      for (int h = 0; h < H; ++h) acc[h] += As[tt * H + h] * hv;
    }
    __syncthreads();
  }
#pragma unroll
  for (int h = 0; h < H; ++h) Mh[(size_t)h * D + j] = acc[h];
}

// ---------------- ctx[h*128+d] = M[h] . Wv[:, h*128+d]
__global__ __launch_bounds__(128) void ctx_kernel(
    const float* __restrict__ Mh, const float* __restrict__ Wv,
    float* __restrict__ ctx) {
  const int h = blockIdx.x, d = threadIdx.x;
  const float* mh = Mh + (size_t)h * D;
  float acc = 0.f;
  for (int i = 0; i < D; ++i) acc += mh[i] * Wv[(size_t)i * D + h * HD + d];
  ctx[h * HD + d] = acc;
}

// ---------------- f32 [R][C] -> bf16 [C][R] tiled transpose+convert
__global__ void tconv_kernel(const float* __restrict__ src,
                             __bf16* __restrict__ dst, int R, int C) {
  __shared__ float tile[32][33];
  int c0 = blockIdx.x * 32, r0 = blockIdx.y * 32;
  int tx = threadIdx.x, ty = threadIdx.y;  // (32,8)
#pragma unroll
  for (int i = 0; i < 4; ++i)
    tile[ty + i * 8][tx] = src[(size_t)(r0 + ty + i * 8) * C + c0 + tx];
  __syncthreads();
#pragma unroll
  for (int i = 0; i < 4; ++i)
    dst[(size_t)(c0 + ty + i * 8) * R + r0 + tx] = (__bf16)tile[tx][ty + i * 8];
}

// ---------------- bf16 WMMA GEMM: C = A[M,K] @ Bt[N,K]^T, fp32 accumulate.
// Double-buffered LDS fed by global_load_async_to_lds_b128 (ASYNCcnt).
// MODE 0: +bias, GELU, bf16 out.   MODE 1: +bias, +resid, f32 out.
template <int MODE>
__global__ __launch_bounds__(256) void gemm_bf16(
    const __bf16* __restrict__ A, const __bf16* __restrict__ Bt,
    const float* __restrict__ bias, const float* __restrict__ resid,
    __bf16* __restrict__ outBF, float* __restrict__ outF,
    int M, int N, int K) {
  constexpr int BK = 32, LDSW = 40, TILE = 128 * LDSW;
  __shared__ __bf16 As[2 * TILE];
  __shared__ __bf16 Bs[2 * TILE];
  const int tid = threadIdx.x;
  const int m0 = blockIdx.y * 128, n0 = blockIdx.x * 128;
  const int w = tid >> 5, lane = tid & 31;
  const int wm = w >> 1, wn = w & 1;     // 4x2 wave grid: 32x64 patch per wave
  const int lr = lane & 15, hi = lane >> 4;
  // per-thread async-copy slots: rows rl and rl+64, 8-half column chunk cl
  const int rl = tid >> 2, cl = (tid & 3) * 8;

  auto issue_tile = [&](int kt, int buf) {
    const int k0 = kt * BK;
    __bf16* a_l = As + buf * TILE;
    __bf16* b_l = Bs + buf * TILE;
    async_ld_b128(A  + (size_t)(m0 + rl)      * K + k0 + cl, a_l + rl * LDSW + cl);
    async_ld_b128(A  + (size_t)(m0 + rl + 64) * K + k0 + cl, a_l + (rl + 64) * LDSW + cl);
    async_ld_b128(Bt + (size_t)(n0 + rl)      * K + k0 + cl, b_l + rl * LDSW + cl);
    async_ld_b128(Bt + (size_t)(n0 + rl + 64) * K + k0 + cl, b_l + (rl + 64) * LDSW + cl);
  };

  const int NT = K / BK;
  issue_tile(0, 0);
  v8f acc[2][4] = {};

  for (int kt = 0; kt < NT; ++kt) {
    const int cur = kt & 1;
    wait_async0();        // tile kt landed in LDS[cur] (this wave's copies)
    __syncthreads();      // all waves landed; all done reading LDS[cur^1]
    if (kt + 1 < NT) issue_tile(kt + 1, cur ^ 1);  // overlaps with compute below

    const __bf16* a_l = As + cur * TILE;
    const __bf16* b_l = Bs + cur * TILE;
    // A frag: lane(lr,hi) holds row lr, K = {hi*8..+7} U {16+hi*8..+7}
    FragBF a[2], b[4];
#pragma unroll
    for (int mi = 0; mi < 2; ++mi) {
      const uint4* p = reinterpret_cast<const uint4*>(a_l + (wm * 32 + mi * 16 + lr) * LDSW);
      a[mi].q[0] = p[hi];
      a[mi].q[1] = p[2 + hi];
    }
    // B frag: lane(lr,hi) holds col lr, K = {hi*8..+7} U {16+hi*8..+7} (Bs is [N][K])
#pragma unroll
    for (int ni = 0; ni < 4; ++ni) {
      const uint4* p = reinterpret_cast<const uint4*>(b_l + (wn * 64 + ni * 16 + lr) * LDSW);
      b[ni].q[0] = p[hi];
      b[ni].q[1] = p[2 + hi];
    }
#pragma unroll
    for (int mi = 0; mi < 2; ++mi)
#pragma unroll
      for (int ni = 0; ni < 4; ++ni)
        acc[mi][ni] = __builtin_amdgcn_wmma_f32_16x16x32_bf16(
            false, a[mi].v, false, b[ni].v, (short)0, acc[mi][ni], false, false);
  }

  // C layout: VGPR r, lanes 0-15 -> M=r, lanes 16-31 -> M=r+8; col = lane%16
#pragma unroll
  for (int mi = 0; mi < 2; ++mi)
#pragma unroll
    for (int ni = 0; ni < 4; ++ni)
#pragma unroll
      for (int r = 0; r < 8; ++r) {
        int row = m0 + wm * 32 + mi * 16 + r + 8 * hi;
        int col = n0 + wn * 64 + ni * 16 + lr;
        size_t off = (size_t)row * N + col;
        float val = acc[mi][ni][r] + bias[col];
        if (MODE == 0) outBF[off] = (__bf16)gelu_tanh(val);
        else           outF[off]  = val + resid[off];
      }
}

extern "C" void kernel_launch(void* const* d_in, const int* in_sizes, int n_in,
                              void* d_out, int out_size, void* d_ws, size_t ws_size,
                              hipStream_t stream) {
  const float* x    = (const float*)d_in[0];
  const float* Wq   = (const float*)d_in[1];
  const float* Wk   = (const float*)d_in[2];
  const float* Wv   = (const float*)d_in[3];
  const float* Wo   = (const float*)d_in[4];
  const float* bo   = (const float*)d_in[5];
  const float* W1   = (const float*)d_in[6];
  const float* b1   = (const float*)d_in[7];
  const float* W2   = (const float*)d_in[8];
  const float* b2   = (const float*)d_in[9];
  const float* ln1s = (const float*)d_in[10];
  const float* ln1b = (const float*)d_in[11];
  const float* ln2s = (const float*)d_in[12];
  const float* ln2b = (const float*)d_in[13];
  float* out = (float*)d_out;

  char* ws = (char*)d_ws;
  size_t off = 0;
  auto alloc = [&](size_t bytes) -> void* {
    void* p = ws + off;
    off += (bytes + 255) & ~(size_t)255;
    return p;
  };
  float*  h1  = (float*) alloc((size_t)T * D * 4);     // LN1(x)
  float*  x2  = (float*) alloc((size_t)T * D * 4);     // x + att_out (residual)
  __bf16* h2  = (__bf16*)alloc((size_t)T * D * 2);     // LN2(x2), bf16
  __bf16* W1t = (__bf16*)alloc((size_t)D * D4 * 2);    // W1^T bf16 [4D][D]
  __bf16* W2t = (__bf16*)alloc((size_t)D * D4 * 2);    // W2^T bf16 [D][4D]
  __bf16* U   = (__bf16*)alloc((size_t)T * D4 * 2);    // gelu(h2@W1+b1)
  float*  q   = (float*) alloc((size_t)D * 4);
  float*  g   = (float*) alloc((size_t)H * D * 4);
  float*  S   = (float*) alloc((size_t)T * H * 4);
  float*  Mh  = (float*) alloc((size_t)H * D * 4);
  float*  ctx = (float*) alloc((size_t)D * 4);
  float*  att = (float*) alloc((size_t)D * 4);
  (void)in_sizes; (void)n_in; (void)out_size; (void)ws_size;

  // h1 = LN1(x)
  ln_kernel<<<T, 256, 0, stream>>>(x, nullptr, ln1s, ln1b, nullptr, h1, nullptr);
  // q = h1[T-1] @ Wq
  matvec_kernel<<<D / 256, 256, 0, stream>>>(h1 + (size_t)(T - 1) * D, Wq, nullptr, q, D, D);
  // g[h] = Wk_head @ q_head
  gvec_kernel<<<H * D / 8, 256, 0, stream>>>(Wk, q, g);
  // S[t][h] = h1[t].g[h] / sqrt(HD)
  scores_kernel<<<T, 512, 0, stream>>>(h1, g, S);
  // softmax over t per head (causal: last row attends to all)
  softmax_kernel<<<H, 256, 0, stream>>>(S);
  // Mh[h] = sum_t S[t][h] * h1[t]
  wsum_kernel<<<D / 256, 256, 0, stream>>>(S, h1, Mh);
  // ctx[h*128+d] = Mh[h] . Wv[:, h*128+d]
  ctx_kernel<<<H, HD, 0, stream>>>(Mh, Wv, ctx);
  // att = ctx @ Wo + bo
  matvec_kernel<<<D / 256, 256, 0, stream>>>(ctx, Wo, bo, att, D, D);
  // x2 = x + att (broadcast); h2 = LN2(x2) in bf16
  ln_kernel<<<T, 256, 0, stream>>>(x, att, ln2s, ln2b, x2, nullptr, h2);
  // weight transposes to bf16
  tconv_kernel<<<dim3(D4 / 32, D / 32), dim3(32, 8), 0, stream>>>(W1, W1t, D, D4);
  tconv_kernel<<<dim3(D / 32, D4 / 32), dim3(32, 8), 0, stream>>>(W2, W2t, D4, D);
  // U = gelu(h2 @ W1 + b1)          [WMMA bf16, async double-buffered]
  gemm_bf16<0><<<dim3(D4 / 128, T / 128), 256, 0, stream>>>(h2, W1t, b1, nullptr, U, nullptr, T, D4, D);
  // out = U @ W2 + b2 + x2          [WMMA bf16, async double-buffered]
  gemm_bf16<1><<<dim3(D / 128, T / 128), 256, 0, stream>>>(U, W2t, b2, x2, nullptr, out, T, D, D4);
}
